// TangetBundle_56315611185441
// MI455X (gfx1250) — matrix operand order repaired and checked by reading, not compiled, not run on hardware
//
#include <hip/hip_runtime.h>
#include <hip/hip_bf16.h>
#include <math.h>

typedef __attribute__((ext_vector_type(16))) __bf16 v16bf;
typedef __attribute__((ext_vector_type(8)))  float  v8f;

#define B_N   2048
#define DY    64
#define NC    8
#define NM    512
#define DZ    64
#define NH    256
#define DHALF 32
#define CM    (NC*NM)      /* 4096 landmarks total */

/* workspace layout (bytes) */
#define OFF_L    0
#define OFF_LSQ  (OFF_L   + CM*DY*2)        /* 524288  */
#define OFF_PSI  (OFF_LSQ + CM*4)           /* 540672  */
#define OFF_PHI  (OFF_PSI + NC*DY*DZ*2)     /* 606208  */
#define OFF_W1   (OFF_PHI + NC*DZ*DY*2)     /* 671744  */
#define OFF_W2   (OFF_W1  + NC*DZ*NH*2)     /* 933888  */

/* LDS row strides (padded to dodge bank conflicts; multiples of 4 floats) */
#define ZS 68
#define YS 68
#define HS 264

static __device__ inline v8f v8zero() {
    v8f c;
#pragma unroll
    for (int i = 0; i < 8; ++i) c[i] = 0.0f;
    return c;
}

static __device__ inline v8f wmma_bf16(v16bf a, v16bf b, v8f c) {
    /* D = A(16x32 bf16) * B(32x16 bf16) + C(16x16 f32) */
    return __builtin_amdgcn_wmma_f32_16x16x32_bf16(false, a, false, b,
                                                   (short)0, c, false, false);
}

/* same-wave LDS producer->consumer: hardware keeps same-wave DS ops in order;
   this only stops the compiler from reordering across the boundary. */
static __device__ inline void syncw() {
    __builtin_amdgcn_wave_barrier();
    asm volatile("" ::: "memory");
}

/* A-operand (16x32, bf16) built from one f32 row in LDS.
   ISA layout: lane M = lane&15, half = lane>>4;
   VGPR v holds K = kb + 8*half + 2*v (+8 if v>=4), and K+1. */
static __device__ inline v16bf afrag_f32(const float* row, int half, int kb) {
    v16bf a;
#pragma unroll
    for (int v = 0; v < 8; ++v) {
        int kp = kb + 8*half + 2*v + ((v >= 4) ? 8 : 0);
        a[2*v]   = (__bf16)row[kp];
        a[2*v+1] = (__bf16)row[kp + 1];
    }
    return a;
}

static __device__ inline v16bf afrag_bf16(const __bf16* row, int half, int kb) {
    v16bf a;
#pragma unroll
    for (int v = 0; v < 8; ++v) {
        int kp = kb + 8*half + 2*v + ((v >= 4) ? 8 : 0);
        a[2*v]   = row[kp];
        a[2*v+1] = row[kp + 1];
    }
    return a;
}

/* B-operand fragment, pre-packed: frag f is 32 lanes x 32 bytes contiguous.
   Lane layout (dense analogue of ISA sparse-B table): N = lane&15,
   element j holds K = 8*(lane>>4) + j (+8 if j>=8) within the 32-K tile, as in afrag. */
static __device__ inline v16bf bfrag(const __bf16* base, int f, int lane) {
    return *(const v16bf*)(base + ((size_t)(f*32 + lane)) * 16);
}

/* ---------------- prep kernels ---------------- */

__global__ void pack_b_kernel(const float* __restrict__ src, __bf16* __restrict__ dst,
                              int fragsPerMat, int Kfrags, int kStride, int nStride,
                              int matSrcStride, int total) {
    int tid = blockIdx.x * blockDim.x + threadIdx.x;
    if (tid >= total) return;
    int perMat = fragsPerMat * 512;
    int mat = tid / perMat;
    int rem = tid - mat * perMat;
    int f   = rem >> 9;
    int r2  = rem & 511;
    int ln  = r2 >> 4;
    int j   = r2 & 15;
    int tn  = f / Kfrags;
    int tk  = f - tn * Kfrags;
    int n   = tn * 16 + (ln & 15);
    int k   = tk * 32 + (ln >> 4) * 8 + j + ((j >= 8) ? 8 : 0);
    dst[tid] = (__bf16)src[(size_t)mat * matSrcStride + (size_t)k * kStride + (size_t)n * nStride];
}

__global__ void lsq_kernel(const float* __restrict__ Lm, float* __restrict__ Lsq) {
    int i = blockIdx.x * blockDim.x + threadIdx.x;
    if (i >= CM) return;
    float s = 0.0f;
#pragma unroll 8
    for (int k = 0; k < DY; ++k) { float v = Lm[i * DY + k]; s += v * v; }
    Lsq[i] = s;
}

/* ---------------- main kernel: one wave32 per 16-row batch tile ---------------- */

__global__ __launch_bounds__(32) void tb_main(
    const float* __restrict__ y_in, const float* __restrict__ t_ptr,
    const int*   __restrict__ ns_ptr,
    const float* __restrict__ psi_b, const float* __restrict__ phi_b,
    const float* __restrict__ b1,
    const __bf16* __restrict__ Lp,  const float* __restrict__ Lsq,
    const __bf16* __restrict__ Psip, const __bf16* __restrict__ Phip,
    const __bf16* __restrict__ W1p,  const __bf16* __restrict__ W2p,
    float* __restrict__ out)
{
    __shared__ float  zq[16 * ZS];     /* z state, f32 */
    __shared__ float  yb[16 * YS];     /* decoded y, f32 */
    __shared__ int    ch_cur[16];
    __shared__ int    ch_new[16];
    __shared__ __bf16 hbuf[16 * HS];   /* MLP hidden, bf16 */

    const int   lane = threadIdx.x;
    const int   mrow = lane & 15;
    const int   half = lane >> 4;
    const int   row0 = blockIdx.x * 16;
    const int   nsteps = ns_ptr[0];
    const float dt = t_ptr[0] / (float)nsteps;

    /* load y_in tile (float4-vectorized, b128 both sides) */
    for (int e = lane; e < 16 * (DY / 4); e += 32) {
        int m = e >> 4, k4 = e & 15;
        float4 v = ((const float4*)y_in)[(size_t)(row0 + m) * (DY / 4) + k4];
        *(float4*)&yb[m * YS + k4 * 4] = v;
    }
    if (lane < 16) ch_cur[lane] = 0;
    syncw();

    /* z0 = y_in @ Psi[0] + psi_b[0] */
    {
        v16bf a0 = afrag_f32(&yb[mrow * YS], half, 0);
        v16bf a1 = afrag_f32(&yb[mrow * YS], half, 32);
        for (int tn = 0; tn < 4; ++tn) {
            v8f C = v8zero();
            C = wmma_bf16(a0, bfrag(Psip, tn * 2 + 0, lane), C);
            C = wmma_bf16(a1, bfrag(Psip, tn * 2 + 1, lane), C);
            float bv = psi_b[tn * 16 + mrow];
#pragma unroll
            for (int r = 0; r < 8; ++r)
                zq[(r + 8 * half) * ZS + tn * 16 + mrow] = C[r] + bv;
        }
    }
    syncw();

    for (int s = 0; s < nsteps; ++s) {
        /* charts present in this tile (wave-uniform mask) */
        int mask = 0;
#pragma unroll
        for (int m = 0; m < 16; ++m) mask |= 1 << ch_cur[m];

        /* ---- MLP: acc = tanh(z@W1[c]+b1[c]) @ W2[c], per-row selected ---- */
        v16bf az0 = afrag_f32(&zq[mrow * ZS], half, 0);
        v16bf az1 = afrag_f32(&zq[mrow * ZS], half, 32);
        float accsel[16];
#pragma unroll
        for (int i = 0; i < 16; ++i) accsel[i] = 0.0f;

        for (int c = 0; c < NC; ++c) {
            if (!((mask >> c) & 1)) continue;
            const __bf16* w1c = W1p + (size_t)c * (DZ * NH);
            for (int tn = 0; tn < NH / 16; ++tn) {
                v8f C = v8zero();
                C = wmma_bf16(az0, bfrag(w1c, tn * 2 + 0, lane), C);
                C = wmma_bf16(az1, bfrag(w1c, tn * 2 + 1, lane), C);
                float bv = b1[c * NH + tn * 16 + mrow];
#pragma unroll
                for (int r = 0; r < 8; ++r)
                    hbuf[(r + 8 * half) * HS + tn * 16 + mrow] = (__bf16)tanhf(C[r] + bv);
            }
            syncw();
            const __bf16* w2c = W2p + (size_t)c * (NH * DHALF);
            for (int tn = 0; tn < 2; ++tn) {
                v8f C = v8zero();
#pragma unroll
                for (int tk = 0; tk < 8; ++tk) {
                    v16bf a = afrag_bf16(&hbuf[mrow * HS], half, tk * 32);
                    C = wmma_bf16(a, bfrag(w2c, tn * 8 + tk, lane), C);
                }
#pragma unroll
                for (int r = 0; r < 8; ++r)
                    if (ch_cur[r + 8 * half] == c) accsel[tn * 8 + r] = C[r];
            }
            syncw();
        }

        /* ---- leapfrog: q += dt*p ; p += dt*acc (each element one lane) ---- */
#pragma unroll
        for (int tn = 0; tn < 2; ++tn) {
#pragma unroll
            for (int r = 0; r < 8; ++r) {
                int m = r + 8 * half, col = tn * 16 + mrow;
                float qo = zq[m * ZS + col];
                float po = zq[m * ZS + DHALF + col];
                zq[m * ZS + col]         = qo + dt * po;
                zq[m * ZS + DHALF + col] = po + dt * accsel[tn * 8 + r];
            }
        }
        syncw();

        /* ---- decode: y = z @ Phi[chart] + phi_b[chart] ---- */
        az0 = afrag_f32(&zq[mrow * ZS], half, 0);
        az1 = afrag_f32(&zq[mrow * ZS], half, 32);
        float ysel[32];
#pragma unroll
        for (int i = 0; i < 32; ++i) ysel[i] = 0.0f;
        for (int c = 0; c < NC; ++c) {
            if (!((mask >> c) & 1)) continue;
            const __bf16* phic = Phip + (size_t)c * (DZ * DY);
            for (int tn = 0; tn < 4; ++tn) {
                v8f C = v8zero();
                C = wmma_bf16(az0, bfrag(phic, tn * 2 + 0, lane), C);
                C = wmma_bf16(az1, bfrag(phic, tn * 2 + 1, lane), C);
                float bv = phi_b[c * DY + tn * 16 + mrow];
#pragma unroll
                for (int r = 0; r < 8; ++r)
                    if (ch_cur[r + 8 * half] == c) ysel[tn * 8 + r] = C[r] + bv;
            }
        }
#pragma unroll
        for (int tn = 0; tn < 4; ++tn)
#pragma unroll
            for (int r = 0; r < 8; ++r)
                yb[(r + 8 * half) * YS + tn * 16 + mrow] = ysel[tn * 8 + r];
        syncw();

        /* ---- nearest landmark: argmin over 4096.
           |y|^2 is constant per row -> argmin key reduces to Lsq - 2*dot. ---- */
        v16bf ay0 = afrag_f32(&yb[mrow * YS], half, 0);
        v16bf ay1 = afrag_f32(&yb[mrow * YS], half, 32);
        float best[8]; int bidx[8];
#pragma unroll
        for (int r = 0; r < 8; ++r) { best[r] = 3.4e38f; bidx[r] = 0; }
#pragma unroll 4
        for (int tn = 0; tn < CM / 16; ++tn) {
            v8f C = v8zero();
            C = wmma_bf16(ay0, bfrag(Lp, tn * 2 + 0, lane), C);
            C = wmma_bf16(ay1, bfrag(Lp, tn * 2 + 1, lane), C);
            float lq = Lsq[tn * 16 + mrow];
            int   ni = tn * 16 + mrow;
#pragma unroll
            for (int r = 0; r < 8; ++r) {
                float d2 = fmaf(-2.0f, C[r], lq);
                if (d2 < best[r]) { best[r] = d2; bidx[r] = ni; }
            }
        }
        /* min-reduce across the 16 lanes of each half (first-index tie-break) */
#pragma unroll
        for (int off = 8; off >= 1; off >>= 1) {
#pragma unroll
            for (int r = 0; r < 8; ++r) {
                float ov = __shfl_xor(best[r], off, 32);
                int   oi = __shfl_xor(bidx[r], off, 32);
                if (ov < best[r] || (ov == best[r] && oi < bidx[r])) { best[r] = ov; bidx[r] = oi; }
            }
        }
        if (mrow == 0) {
#pragma unroll
            for (int r = 0; r < 8; ++r) ch_new[r + 8 * half] = bidx[r] / NM;
        }
        syncw();

        /* ---- re-encode rows whose chart changed: z = y @ Psi[new] + psi_b ---- */
        int cmask = 0;
#pragma unroll
        for (int m = 0; m < 16; ++m)
            if (ch_new[m] != ch_cur[m]) cmask |= 1 << ch_new[m];
        for (int c = 0; c < NC; ++c) {
            if (!((cmask >> c) & 1)) continue;
            const __bf16* psic = Psip + (size_t)c * (DY * DZ);
            for (int tn = 0; tn < 4; ++tn) {
                v8f C = v8zero();
                C = wmma_bf16(ay0, bfrag(psic, tn * 2 + 0, lane), C);
                C = wmma_bf16(ay1, bfrag(psic, tn * 2 + 1, lane), C);
                float bv = psi_b[c * DZ + tn * 16 + mrow];
#pragma unroll
                for (int r = 0; r < 8; ++r) {
                    int m = r + 8 * half;
                    if (ch_new[m] == c && ch_new[m] != ch_cur[m])
                        zq[m * ZS + tn * 16 + mrow] = C[r] + bv;
                }
            }
        }
        syncw();
        if (lane < 16) ch_cur[lane] = ch_new[lane];
        syncw();
    }

    /* ---- final decode: y = z @ Phi[chart] + phi_b[chart] ---- */
    {
        int mask = 0;
#pragma unroll
        for (int m = 0; m < 16; ++m) mask |= 1 << ch_cur[m];
        v16bf az0 = afrag_f32(&zq[mrow * ZS], half, 0);
        v16bf az1 = afrag_f32(&zq[mrow * ZS], half, 32);
        float ysel[32];
#pragma unroll
        for (int i = 0; i < 32; ++i) ysel[i] = 0.0f;
        for (int c = 0; c < NC; ++c) {
            if (!((mask >> c) & 1)) continue;
            const __bf16* phic = Phip + (size_t)c * (DZ * DY);
            for (int tn = 0; tn < 4; ++tn) {
                v8f C = v8zero();
                C = wmma_bf16(az0, bfrag(phic, tn * 2 + 0, lane), C);
                C = wmma_bf16(az1, bfrag(phic, tn * 2 + 1, lane), C);
                float bv = phi_b[c * DY + tn * 16 + mrow];
#pragma unroll
                for (int r = 0; r < 8; ++r)
                    if (ch_cur[r + 8 * half] == c) ysel[tn * 8 + r] = C[r] + bv;
            }
        }
#pragma unroll
        for (int tn = 0; tn < 4; ++tn)
#pragma unroll
            for (int r = 0; r < 8; ++r)
                yb[(r + 8 * half) * YS + tn * 16 + mrow] = ysel[tn * 8 + r];
        syncw();
        for (int e = lane; e < 16 * (DY / 4); e += 32) {
            int m = e >> 4, k4 = e & 15;
            float4 v = *(const float4*)&yb[m * YS + k4 * 4];
            ((float4*)out)[(size_t)(row0 + m) * (DY / 4) + k4] = v;
        }
    }
}

/* ---------------- host launcher ---------------- */

extern "C" void kernel_launch(void* const* d_in, const int* in_sizes, int n_in,
                              void* d_out, int out_size, void* d_ws, size_t ws_size,
                              hipStream_t stream) {
    (void)in_sizes; (void)n_in; (void)out_size; (void)ws_size;
    const float* y_in   = (const float*)d_in[0];
    const float* t      = (const float*)d_in[1];
    const float* Psi    = (const float*)d_in[2];
    const float* psi_b  = (const float*)d_in[3];
    const float* Phi    = (const float*)d_in[4];
    const float* phi_b  = (const float*)d_in[5];
    const float* W1     = (const float*)d_in[6];
    const float* b1     = (const float*)d_in[7];
    const float* W2     = (const float*)d_in[8];
    const float* Lm     = (const float*)d_in[9];
    const int*   nsteps = (const int*)d_in[10];

    char* ws = (char*)d_ws;
    __bf16* Lp   = (__bf16*)(ws + OFF_L);
    float*  Lsq  = (float*) (ws + OFF_LSQ);
    __bf16* Psip = (__bf16*)(ws + OFF_PSI);
    __bf16* Phip = (__bf16*)(ws + OFF_PHI);
    __bf16* W1p  = (__bf16*)(ws + OFF_W1);
    __bf16* W2p  = (__bf16*)(ws + OFF_W2);

    /* pack B operands: (src, dst, fragsPerMat, Kfrags, kStride, nStride, matStride, total) */
    pack_b_kernel<<<(CM*DY + 255) / 256, 256, 0, stream>>>(Lm,  Lp,   (CM/16)*2, 2, 1,   DY, 0,       CM*DY);
    pack_b_kernel<<<(NC*DY*DZ + 255) / 256, 256, 0, stream>>>(Psi, Psip, 8,        2, DZ,  1,  DY*DZ,   NC*DY*DZ);
    pack_b_kernel<<<(NC*DZ*DY + 255) / 256, 256, 0, stream>>>(Phi, Phip, 8,        2, DY,  1,  DZ*DY,   NC*DZ*DY);
    pack_b_kernel<<<(NC*DZ*NH + 255) / 256, 256, 0, stream>>>(W1,  W1p,  (NH/16)*2, 2, NH,  1,  DZ*NH,   NC*DZ*NH);
    pack_b_kernel<<<(NC*NH*DHALF + 255) / 256, 256, 0, stream>>>(W2, W2p, 2*(NH/32), 8, DHALF, 1, NH*DHALF, NC*NH*DHALF);
    lsq_kernel<<<(CM + 255) / 256, 256, 0, stream>>>(Lm, Lsq);

    tb_main<<<B_N / 16, 32, 0, stream>>>(y_in, t, nsteps, psi_b, phi_b, b1,
                                         Lp, Lsq, Psip, Phip, W1p, W2p,
                                         (float*)d_out);
}
